// PoseRDE_53678501266110
// MI455X (gfx1250) — matrix operand order repaired and checked by reading, not compiled, not run on hardware
//
#include <hip/hip_runtime.h>

// ---------------------------------------------------------------------------
// PoseRDE on MI455X (gfx1250). Sequential RK scan; all matmuls via
// v_wmma_f32_16x16x32_bf16 (bf16 A/B, f32 accumulate). Weights repacked once
// per call into [N][Kpad=160] bf16 with zero padding so pads contribute 0.
// ---------------------------------------------------------------------------

typedef __bf16 bf16_t;
typedef __attribute__((ext_vector_type(16))) __bf16 v16bf;
typedef __attribute__((ext_vector_type(8)))  float  v8f;

#define WMMA_BF16(A_, B_, C_) \
  __builtin_amdgcn_wmma_f32_16x16x32_bf16(false, (A_), false, (B_), (short)0, (C_), false, false)

// Problem constants
#define NB   32     // batch
#define LSEQ 41     // SEQ-1
#define C0V  17
#define CHV  153
#define HIDV 152
#define KP   160    // padded K/N (multiple of 32)
#define ROWSTRIDE 160
#define ZELEMS (NB * KP)        // 5120
#define NSTEP 40

// -------------------- fragment loaders (contiguous k = 16h + s order) -----------------
// A 16x32 tile: lane<16 holds M=lane, K 0..15; lane>=16 holds
// M=lane-16, K 16..31. base is row-major [M][KP] bf16.
__device__ __forceinline__ v16bf load_afrag(const bf16_t* base, int mrow0, int lane, int kstep) {
  int m  = mrow0 + (lane & 15);
  int kb = kstep * 32 + ((lane & 16) ? 16 : 0);
  const bf16_t* p = base + m * ROWSTRIDE + kb;
  v16bf a;
#pragma unroll
  for (int i = 0; i < 16; ++i) { a[i] = p[i]; }
  return a;
}

// B 32x16 tile: lane l holds column N=l&15, halves = K (l<16 ? 0..15 : 16..31)
// Wb is [N][KP] bf16 (i.e., W[n][k] for out[m,n]=sum_k A[m,k]W[n,k]).
__device__ __forceinline__ v16bf load_bfrag(const bf16_t* Wb, int n0, int lane, int kstep) {
  int n  = n0 + (lane & 15);
  int kb = kstep * 32 + ((lane & 16) ? 16 : 0);
  return *(const v16bf*)(Wb + n * ROWSTRIDE + kb);
}

// -------------------- weight packing -----------------------------------------
__global__ void prep_pack_small(const float* W_c0, const float* W_c1, const float* W_init,
                                const float* W_r1, const float* b_c0, const float* b_c1,
                                const float* b_init, const float* b_cf,
                                bf16_t* wc0b, bf16_t* wc1b, bf16_t* winitb, bf16_t* wr1b,
                                float* bc0p, float* bc1p, float* binitp, float* bcfp) {
  int idx = blockIdx.x * 256 + threadIdx.x;
  if (idx < 25600) {
    int n = idx / KP, k = idx % KP;
    wc0b[idx] = (n < HIDV && k < HIDV) ? (bf16_t)W_c0[n * HIDV + k] : (bf16_t)0.f;
  } else if (idx < 51200) {
    int j = idx - 25600; int n = j / KP, k = j % KP;
    wc1b[j] = (n < HIDV && k < HIDV) ? (bf16_t)W_c1[n * HIDV + k] : (bf16_t)0.f;
  } else if (idx < 76800) {
    int j = idx - 51200; int n = j / KP, k = j % KP;
    winitb[j] = (n < HIDV && k < CHV) ? (bf16_t)W_init[n * CHV + k] : (bf16_t)0.f;
  } else if (idx < 97280) {
    int j = idx - 76800; int n = j / KP, k = j % KP;   // 128 rows
    wr1b[j] = (k < HIDV) ? (bf16_t)W_r1[n * HIDV + k] : (bf16_t)0.f;
  } else if (idx < 97440) {
    int k = idx - 97280; bc0p[k] = (k < HIDV) ? b_c0[k] : 0.f;
  } else if (idx < 97600) {
    int k = idx - 97440; bc1p[k] = (k < HIDV) ? b_c1[k] : 0.f;
  } else if (idx < 97760) {
    int k = idx - 97600; binitp[k] = (k < HIDV) ? b_init[k] : 0.f;
  } else if (idx < 122080) {
    int j = idx - 97760; int hh = j / KP, c = j % KP;
    bcfp[j] = (c < CHV) ? b_cf[hh * CHV + c] : 0.f;
  }
}

__global__ void prep_wcf(const float* W_cf, bf16_t* wcfb) {
  int idx = blockIdx.x * 256 + threadIdx.x;
  if (idx >= HIDV * KP * KP) return;
  int hh = idx / (KP * KP); int rem = idx % (KP * KP);
  int c = rem / KP, k = rem % KP;
  wcfb[idx] = (c < CHV && k < HIDV) ? (bf16_t)W_cf[(hh * CHV + c) * HIDV + k] : (bf16_t)0.f;
}

// -------------------- log-signature path -------------------------------------
// Produces d0/d1 = path increments ([32][160] f32, zero padded) and path0
// (bf16 [32][160], zero padded) for the z0 GEMM.
__global__ void prep_path(const float* fv, const float* fi, const float* ts,
                          bf16_t* path0b, float* d0g, float* d1g) {
  int b = blockIdx.x; int t = threadIdx.x; // 128 threads
  __shared__ float xls[LSEQ][C0V];
  __shared__ float dls[20][C0V];
  __shared__ float cpls[20][C0V];
  __shared__ float Els[C0V][C0V];
  __shared__ float rows[2][KP];
  for (int e = t; e < LSEQ * C0V; e += blockDim.x) {
    int l = e / C0V, i = e % C0V;
    float v;
    if (i == 0)      v = ts[b * 42 + l + 1];
    else if (i <= 8) v = fv[(b * LSEQ + l) * 8 + (i - 1)];
    else             v = fi[(b * LSEQ + l) * 8 + (i - 9)];
    xls[l][i] = v;
  }
  __syncthreads();
  for (int w = 0; w < 2; ++w) {
    for (int i = t; i < C0V; i += blockDim.x) {
      float run = 0.f;
      for (int l = 0; l < 20; ++l) {
        float dv = xls[w * 20 + l + 1][i] - xls[w * 20 + l][i];
        dls[l][i] = dv;
        cpls[l][i] = run;   // exclusive prefix (cumsum - d)
        run += dv;
      }
    }
    __syncthreads();
    for (int e = t; e < C0V * C0V; e += blockDim.x) {
      int i = e / C0V, j = e % C0V;
      float s = 0.f;
      for (int l = 0; l < 20; ++l) s += cpls[l][i] * dls[l][j];
      Els[i][j] = s;
    }
    __syncthreads();
    for (int col = t; col < KP; col += blockDim.x) {
      float v = 0.f;
      if (col < C0V) v = xls[w * 20 + 20][col] - xls[w * 20][col];
      else if (col < CHV) {
        int p = col - C0V;                       // triu (k=1) row-major pair index
        int i = 0, acc = 0;
        while (p >= acc + (16 - i)) { acc += 16 - i; ++i; }
        int j = i + 1 + (p - acc);
        v = 0.5f * (Els[i][j] - Els[j][i]);
      }
      rows[w][col] = v;
    }
    __syncthreads();
  }
  for (int col = t; col < KP; col += blockDim.x) {
    d0g[b * KP + col] = rows[0][col];
    d1g[b * KP + col] = rows[1][col];
    float p0 = (col < C0V) ? xls[0][col] : 0.f;
    path0b[b * KP + col] = (bf16_t)p0;
  }
}

// -------------------- z0 = tanh(path0 @ Winit^T + binit) ----------------------
__global__ __launch_bounds__(320) void z0_kernel(const bf16_t* path0b, const bf16_t* winitb,
                                                 const float* binitp, float* zbuf,
                                                 bf16_t* hib, float* dout_z0) {
  int tid = threadIdx.x; int wave = tid >> 5; int lane = tid & 31; // 10 waves
  int n0 = wave * 16;
  v8f acc0 = {0,0,0,0,0,0,0,0}, acc1 = {0,0,0,0,0,0,0,0};
#pragma unroll
  for (int k = 0; k < 5; ++k) {
    v16bf bfrag = load_bfrag(winitb, n0, lane, k);
    v16bf a0 = load_afrag(path0b, 0,  lane, k);
    v16bf a1 = load_afrag(path0b, 16, lane, k);
    acc0 = WMMA_BF16(a0, bfrag, acc0);
    acc1 = WMMA_BF16(a1, bfrag, acc1);
  }
  int n = n0 + (lane & 15);
  float bias = binitp[n];
#pragma unroll
  for (int r = 0; r < 8; ++r) {
    int row0 = r + ((lane & 16) ? 8 : 0);
    float v0 = tanhf(acc0[r] + bias);
    float v1 = tanhf(acc1[r] + bias);
    if (n >= HIDV) { v0 = 0.f; v1 = 0.f; }     // pads exactly zero
    zbuf[row0 * KP + n] = v0;           zbuf[(row0 + 16) * KP + n] = v1;
    hib[row0 * KP + n] = (bf16_t)v0;    hib[(row0 + 16) * KP + n] = (bf16_t)v1;
    if (n < HIDV) { dout_z0[row0 * HIDV + n] = v0; dout_z0[(row0 + 16) * HIDV + n] = v1; }
  }
}

// -------------------- stage A: RK stage input + h1,h2 MLP ---------------------
__global__ __launch_bounds__(320) void stageA(int step, int stage, const float* ts,
                                              float* kbuf, float* zbuf, bf16_t* hib,
                                              const bf16_t* wc0b, const float* bc0p,
                                              const bf16_t* wc1b, const float* bc1p,
                                              bf16_t* h2b) {
  __shared__ bf16_t zin[ZELEMS];
  __shared__ bf16_t h1s[ZELEMS];
  int tid = threadIdx.x;
  const float* k1 = kbuf;              const float* k2 = kbuf + ZELEMS;
  const float* k3 = kbuf + 2 * ZELEMS; const float* k4 = kbuf + 3 * ZELEMS;
  float t0 = ts[step + 1] - ts[0];
  float t1 = ts[step + 2] - ts[0];
  float h  = t1 - t0;
  for (int e = tid; e < ZELEMS; e += 320) {
    int col = e % KP;
    float v = 0.f;
    if (col < HIDV) {
      float z = zbuf[e];
      if (stage == 0) {
        if (step > 0) {                          // finalize previous step
          float hp = ts[step + 1] - ts[step];
          z += hp * (k1[e] + 3.f * (k2[e] + k3[e]) + k4[e]) * 0.125f;
          zbuf[e] = z;
        }
        v = z;
      } else if (stage == 1) v = z + h * k1[e] * (1.f / 3.f);
      else if (stage == 2)   v = z + h * (k2[e] - k1[e] * (1.f / 3.f));
      else                   v = z + h * (k1[e] - k2[e] + k3[e]);
    }
    if (stage == 0 && step > 0) hib[step * ZELEMS + e] = (bf16_t)v;  // h_i[step]
    zin[e] = (bf16_t)v;
  }
  __syncthreads();
  int wave = tid >> 5, lane = tid & 31; int n0 = wave * 16;
  // layer 1: h1 = tanh(zin @ Wc0^T + bc0)
  {
    v8f acc0 = {0,0,0,0,0,0,0,0}, acc1 = {0,0,0,0,0,0,0,0};
#pragma unroll
    for (int k = 0; k < 5; ++k) {
      v16bf bfrag = load_bfrag(wc0b, n0, lane, k);
      v16bf a0 = load_afrag(zin, 0,  lane, k);
      v16bf a1 = load_afrag(zin, 16, lane, k);
      acc0 = WMMA_BF16(a0, bfrag, acc0);
      acc1 = WMMA_BF16(a1, bfrag, acc1);
    }
    int n = n0 + (lane & 15);
    float bias = bc0p[n];
#pragma unroll
    for (int r = 0; r < 8; ++r) {
      int row0 = r + ((lane & 16) ? 8 : 0);
      h1s[row0 * KP + n]        = (bf16_t)tanhf(acc0[r] + bias);
      h1s[(row0 + 16) * KP + n] = (bf16_t)tanhf(acc1[r] + bias);
    }
  }
  __syncthreads();
  // layer 2: h2 = tanh(h1 @ Wc1^T + bc1) -> global
  {
    v8f acc0 = {0,0,0,0,0,0,0,0}, acc1 = {0,0,0,0,0,0,0,0};
#pragma unroll
    for (int k = 0; k < 5; ++k) {
      v16bf bfrag = load_bfrag(wc1b, n0, lane, k);
      v16bf a0 = load_afrag(h1s, 0,  lane, k);
      v16bf a1 = load_afrag(h1s, 16, lane, k);
      acc0 = WMMA_BF16(a0, bfrag, acc0);
      acc1 = WMMA_BF16(a1, bfrag, acc1);
    }
    int n = n0 + (lane & 15);
    float bias = bc1p[n];
#pragma unroll
    for (int r = 0; r < 8; ++r) {
      int row0 = r + ((lane & 16) ? 8 : 0);
      h2b[row0 * KP + n]        = (bf16_t)tanhf(acc0[r] + bias);
      h2b[(row0 + 16) * KP + n] = (bf16_t)tanhf(acc1[r] + bias);
    }
  }
}

// -------------------- stage B: k[b,h] = sum_c tanh((h2 Wcf^T)[b,hc]+b)*Xd[b,c]
// One workgroup per h (152 blocks, 4 waves). Never materializes the (32,23256)
// intermediate; tanh + contraction fused in registers; deterministic reduce.
__global__ __launch_bounds__(128) void stageB(int step, int stage, const float* ts,
                                              const bf16_t* h2b, const bf16_t* wcfb,
                                              const float* bcfp, const float* d0g,
                                              const float* d1g, float* kout) {
  int hidx = blockIdx.x;
  int tid = threadIdx.x; int wave = tid >> 5, lane = tid & 31;
  __shared__ float xt[KP * NB];       // xt[c*32 + b]
  __shared__ float wpart[4][NB];
  float t0 = ts[step + 1] - ts[0];
  float t1 = ts[step + 2] - ts[0];
  float h  = t1 - t0;
  float st = (stage == 0) ? t0 : (stage == 1) ? t0 + h * (1.f / 3.f)
           : (stage == 2) ? t0 + h * (2.f / 3.f) : t1;
  const float* xd = (st >= 1.0f) ? d1g : d0g;   // i = clip(floor(st),0,1)
  for (int e = tid; e < KP * NB; e += 128) {
    int c = e >> 5, b = e & 31;
    xt[e] = xd[b * KP + c];
  }
  __syncthreads();
  v16bf A0[5], A1[5];
#pragma unroll
  for (int k = 0; k < 5; ++k) { A0[k] = load_afrag(h2b, 0, lane, k); A1[k] = load_afrag(h2b, 16, lane, k); }
  float g0[8], g1[8];
#pragma unroll
  for (int r = 0; r < 8; ++r) { g0[r] = 0.f; g1[r] = 0.f; }
  const bf16_t* Wh = wcfb + (size_t)hidx * KP * KP;
  int ntCount = (wave < 2) ? 3 : 2;             // 10 n-tiles over 4 waves
  int rb = (lane & 16) ? 8 : 0;
  for (int q = 0; q < ntCount; ++q) {
    int ntile = wave + q * 4;
    v8f acc0 = {0,0,0,0,0,0,0,0}, acc1 = {0,0,0,0,0,0,0,0};
#pragma unroll
    for (int k = 0; k < 5; ++k) {
      v16bf bfrag = load_bfrag(Wh, ntile * 16, lane, k);
      acc0 = WMMA_BF16(A0[k], bfrag, acc0);
      acc1 = WMMA_BF16(A1[k], bfrag, acc1);
    }
    int c = ntile * 16 + (lane & 15);
    float bias = bcfp[hidx * KP + c];
#pragma unroll
    for (int r = 0; r < 8; ++r) {
      g0[r] += tanhf(acc0[r] + bias) * xt[c * 32 + rb + r];
      g1[r] += tanhf(acc1[r] + bias) * xt[c * 32 + 16 + rb + r];
    }
  }
  // butterfly within each 16-lane half (all 16 lanes share the same row)
#pragma unroll
  for (int r = 0; r < 8; ++r) {
    float s0 = g0[r], s1 = g1[r];
#pragma unroll
    for (int m = 1; m < 16; m <<= 1) { s0 += __shfl_xor(s0, m, 32); s1 += __shfl_xor(s1, m, 32); }
    g0[r] = s0; g1[r] = s1;
  }
  if ((lane & 15) == 0) {
#pragma unroll
    for (int r = 0; r < 8; ++r) { wpart[wave][rb + r] = g0[r]; wpart[wave][16 + rb + r] = g1[r]; }
  }
  __syncthreads();
  if (tid < NB) {
    float s = wpart[0][tid] + wpart[1][tid] + wpart[2][tid] + wpart[3][tid];
    kout[tid * KP + hidx] = s;
  }
}

// -------------------- final z update (h_i[40]) --------------------------------
__global__ __launch_bounds__(320) void finalZ(const float* ts, const float* kbuf,
                                              const float* zbuf, bf16_t* hib) {
  int tid = threadIdx.x;
  const float* k1 = kbuf;              const float* k2 = kbuf + ZELEMS;
  const float* k3 = kbuf + 2 * ZELEMS; const float* k4 = kbuf + 3 * ZELEMS;
  float hp = ts[41] - ts[40];
  for (int e = tid; e < ZELEMS; e += 320) {
    int col = e % KP;
    float v = 0.f;
    if (col < HIDV)
      v = zbuf[e] + hp * (k1[e] + 3.f * (k2[e] + k3[e]) + k4[e]) * 0.125f;
    hib[NSTEP * ZELEMS + e] = (bf16_t)v;
  }
}

// -------------------- readout layer 1 (leaky relu) ----------------------------
__global__ __launch_bounds__(128) void readout1(const bf16_t* hib, const bf16_t* wr1b,
                                                const float* br1, bf16_t* hrb) {
  int mt = blockIdx.x;                  // 82 row tiles of 16 over 1312 rows
  int tid = threadIdx.x; int wave = tid >> 5, lane = tid & 31;
  v16bf A[5];
#pragma unroll
  for (int k = 0; k < 5; ++k) A[k] = load_afrag(hib, mt * 16, lane, k);
  for (int q = 0; q < 2; ++q) {
    int n0 = (wave + q * 4) * 16;
    v8f acc = {0,0,0,0,0,0,0,0};
#pragma unroll
    for (int k = 0; k < 5; ++k) {
      v16bf bfrag = load_bfrag(wr1b, n0, lane, k);
      acc = WMMA_BF16(A[k], bfrag, acc);
    }
    int n = n0 + (lane & 15);
    float bias = br1[n];
#pragma unroll
    for (int r = 0; r < 8; ++r) {
      int row = mt * 16 + r + ((lane & 16) ? 8 : 0);
      float v = acc[r] + bias;
      v = (v > 0.f) ? v : 0.1f * v;
      hrb[row * 128 + n] = (bf16_t)v;
    }
  }
}

// -------------------- readout layer 2 (6-wide, scalar) ------------------------
__global__ void readout2(const bf16_t* hrb, const float* W_r2, const float* b_r2, float* dout) {
  int idx = blockIdx.x * blockDim.x + threadIdx.x;
  if (idx >= LSEQ * NB * 6) return;
  int row = idx / 6, o = idx % 6;       // row = t*32 + b
  float s = b_r2[o];
#pragma unroll 16
  for (int k = 0; k < 128; ++k) s += (float)hrb[row * 128 + k] * W_r2[o * 128 + k];
  int b = row & 31, t = row >> 5;
  dout[(b * LSEQ + t) * 6 + o] = s;
}

// -------------------- workspace layout (bytes) --------------------------------
#define OFF_WC0B  ((size_t)0)         // 160*160 bf16
#define OFF_WC1B  ((size_t)51200)
#define OFF_WINIT ((size_t)102400)
#define OFF_WR1B  ((size_t)153600)    // 128*160 bf16
#define OFF_WCFB  ((size_t)194560)    // 152*160*160 bf16 (7.78 MB, L2-resident)
#define OFF_BC0P  ((size_t)7976960)
#define OFF_BC1P  ((size_t)7977600)
#define OFF_BINIT ((size_t)7978240)
#define OFF_BCFP  ((size_t)7978880)   // 152*160 f32
#define OFF_P0B   ((size_t)8076160)   // 32*160 bf16
#define OFF_D0    ((size_t)8086400)   // 32*160 f32
#define OFF_D1    ((size_t)8106880)
#define OFF_Z     ((size_t)8127360)   // 32*160 f32
#define OFF_H2    ((size_t)8147840)   // 32*160 bf16
#define OFF_K     ((size_t)8158080)   // 4 * 32*160 f32
#define OFF_HIB   ((size_t)8240000)   // 41*32*160 bf16
#define OFF_HRB   ((size_t)8659840)   // 1312*128 bf16  (total ~9.0 MB)

extern "C" void kernel_launch(void* const* d_in, const int* in_sizes, int n_in,
                              void* d_out, int out_size, void* d_ws, size_t ws_size,
                              hipStream_t stream) {
  (void)in_sizes; (void)n_in; (void)out_size; (void)ws_size;
  const float* fv     = (const float*)d_in[0];
  const float* fi     = (const float*)d_in[1];
  const float* ts     = (const float*)d_in[2];
  const float* W_init = (const float*)d_in[3];
  const float* b_init = (const float*)d_in[4];
  const float* W_c0   = (const float*)d_in[5];
  const float* b_c0   = (const float*)d_in[6];
  const float* W_c1   = (const float*)d_in[7];
  const float* b_c1   = (const float*)d_in[8];
  const float* W_cf   = (const float*)d_in[9];
  const float* b_cf   = (const float*)d_in[10];
  const float* W_r1   = (const float*)d_in[11];
  const float* b_r1   = (const float*)d_in[12];
  const float* W_r2   = (const float*)d_in[13];
  const float* b_r2   = (const float*)d_in[14];
  char* ws = (char*)d_ws;
  bf16_t* wc0b   = (bf16_t*)(ws + OFF_WC0B);
  bf16_t* wc1b   = (bf16_t*)(ws + OFF_WC1B);
  bf16_t* winitb = (bf16_t*)(ws + OFF_WINIT);
  bf16_t* wr1b   = (bf16_t*)(ws + OFF_WR1B);
  bf16_t* wcfb   = (bf16_t*)(ws + OFF_WCFB);
  float*  bc0p   = (float*)(ws + OFF_BC0P);
  float*  bc1p   = (float*)(ws + OFF_BC1P);
  float*  binitp = (float*)(ws + OFF_BINIT);
  float*  bcfp   = (float*)(ws + OFF_BCFP);
  bf16_t* path0b = (bf16_t*)(ws + OFF_P0B);
  float*  d0g    = (float*)(ws + OFF_D0);
  float*  d1g    = (float*)(ws + OFF_D1);
  float*  zbuf   = (float*)(ws + OFF_Z);
  bf16_t* h2b    = (bf16_t*)(ws + OFF_H2);
  float*  kbuf   = (float*)(ws + OFF_K);
  bf16_t* hib    = (bf16_t*)(ws + OFF_HIB);
  bf16_t* hrb    = (bf16_t*)(ws + OFF_HRB);
  float*  dout   = (float*)d_out;

  prep_pack_small<<<477, 256, 0, stream>>>(W_c0, W_c1, W_init, W_r1, b_c0, b_c1, b_init, b_cf,
                                           wc0b, wc1b, winitb, wr1b, bc0p, bc1p, binitp, bcfp);
  prep_wcf<<<15200, 256, 0, stream>>>(W_cf, wcfb);
  prep_path<<<32, 128, 0, stream>>>(fv, fi, ts, path0b, d0g, d1g);
  z0_kernel<<<1, 320, 0, stream>>>(path0b, winitb, binitp, zbuf, hib, dout + LSEQ * NB * 6);

  for (int i = 0; i < NSTEP; ++i) {
    for (int s = 0; s < 4; ++s) {
      stageA<<<1, 320, 0, stream>>>(i, s, ts, kbuf, zbuf, hib, wc0b, bc0p, wc1b, bc1p, h2b);
      stageB<<<HIDV, 128, 0, stream>>>(i, s, ts, h2b, wcfb, bcfp, d0g, d1g, kbuf + s * ZELEMS);
    }
  }
  finalZ<<<1, 320, 0, stream>>>(ts, kbuf, zbuf, hib);
  readout1<<<82, 128, 0, stream>>>(hib, wr1b, b_r1, hrb);
  readout2<<<31, 256, 0, stream>>>(hrb, W_r2, b_r2, dout);
}
